// SharedBlockGRU_19636590477500
// MI455X (gfx1250) — matrix operand order repaired and checked by reading, not compiled, not run on hardware
//
#include <hip/hip_runtime.h>

#define M_TOTAL   4096
#define SH        512
#define DIN       512
#define T_N       8
#define G3        1536
#define KDIM      512

typedef __attribute__((ext_vector_type(16))) _Float16 v16h;
typedef __attribute__((ext_vector_type(8)))  float    v8f;

// ---------- plain f32 -> f16 convert (weights; 1:1 layout) ----------
__global__ void cvt_f16_kernel(const float* in, unsigned short* out, int n) {
  int i = blockIdx.x * blockDim.x + threadIdx.x;
  if (i < n) ((_Float16*)out)[i] = (_Float16)in[i];
}

// ---------- f32 -> f16 convert + pack into A-fragment-major layout ----------
// A frag (16-bit 16x32): lane half lh holds K run
// [kb+16*lh, +16) (elems 0..15), matching contiguous B loads.
// Packed layout: [tile][kk][lane][e], 16 f16 contiguous per lane.
__global__ void cvt_pack_a_kernel(const float* in, unsigned short* outp, int total) {
  int i = blockIdx.x * blockDim.x + threadIdx.x;   // i = m*512 + k
  if (i >= total) return;
  int m = i >> 9;
  int k = i & 511;
  int tile = m >> 4, laneM = m & 15;
  int kk = k >> 5, rem = k & 31;
  int lh, e;
  lh = rem >> 4; e = rem & 15;
  int lane = lh * 16 + laneM;
  size_t dst = (((size_t)tile * 16 + kk) * 32 + lane) * 16 + e;
  ((_Float16*)outp)[dst] = (_Float16)in[i];
}

// ---------- h_read = hh @ w_read (4096x512 * 512x16) ----------
__global__ void hread_kernel(const float* h, const float* wread, float* hread) {
  int idx = blockIdx.x * blockDim.x + threadIdx.x;   // 65536 threads
  int m = idx >> 4, k = idx & 15;
  const float* row = h + (size_t)m * SH;
  float acc = 0.f;
  for (int s = 0; s < SH; ++s) acc += row[s] * wread[s * 16 + k];
  hread[idx] = acc;
}

// ---------- fused grouped-GEMM + GRU epilogue ----------
// 1D grid: bid = (tile * T_N + t) * 8 + sq ; block = 128 (4 waves)
__global__ void SharedBlockGRU_19636590477500_kernel(
    const unsigned short* xp_, const unsigned short* hp_,
    const unsigned short* Wihb_, const unsigned short* Whhb_,
    const float* bih, const float* bhh,
    const float* hf32, float* hnext) {
  const _Float16* xp   = (const _Float16*)xp_;
  const _Float16* hp   = (const _Float16*)hp_;
  const _Float16* Wihb = (const _Float16*)Wihb_;
  const _Float16* Whhb = (const _Float16*)Whhb_;

  const int bid   = blockIdx.x;
  const int sq    = bid & 7;              // s-quad (4 tiles of 16)
  const int t     = (bid >> 3) & 7;
  const int tile  = bid >> 6;             // m-tile
  const int wave  = threadIdx.x >> 5;
  const int lane  = threadIdx.x & 31;
  const int laneM = lane & 15;
  const int lh    = lane >> 4;
  const int m0    = tile * 16;
  const int s0    = (sq * 4 + wave) * 16;

  const _Float16* ax_base = xp + (((size_t)tile * 16) * 32 + lane) * 16;
  const _Float16* ah_base = hp + (((size_t)tile * 16) * 32 + lane) * 16;

  const size_t wmat = (size_t)G3 * KDIM;
  const _Float16* br_ih = Wihb + (size_t)t * wmat + (size_t)(s0 + laneM) * KDIM + lh * 16;
  const _Float16* bz_ih = br_ih + (size_t)SH * KDIM;
  const _Float16* bn_ih = bz_ih + (size_t)SH * KDIM;
  const _Float16* br_hh = Whhb + (size_t)t * wmat + (size_t)(s0 + laneM) * KDIM + lh * 16;
  const _Float16* bz_hh = br_hh + (size_t)SH * KDIM;
  const _Float16* bn_hh = bz_hh + (size_t)SH * KDIM;

  v8f aRX = {};
  v8f aZX = {};
  v8f aNX = {};
  v8f aRH = {};
  v8f aZH = {};
  v8f aNH = {};

  for (int kk = 0; kk < KDIM / 32; ++kk) {
    const int kb = kk * 32;
    v16h ax = *(const v16h*)(ax_base + (size_t)kk * 512);
    v16h ah = *(const v16h*)(ah_base + (size_t)kk * 512);
    v16h b;
    b = *(const v16h*)(br_ih + kb);
    aRX = __builtin_amdgcn_wmma_f32_16x16x32_f16(false, ax, false, b, (short)0, aRX, false, false);
    b = *(const v16h*)(bz_ih + kb);
    aZX = __builtin_amdgcn_wmma_f32_16x16x32_f16(false, ax, false, b, (short)0, aZX, false, false);
    b = *(const v16h*)(bn_ih + kb);
    aNX = __builtin_amdgcn_wmma_f32_16x16x32_f16(false, ax, false, b, (short)0, aNX, false, false);
    b = *(const v16h*)(br_hh + kb);
    aRH = __builtin_amdgcn_wmma_f32_16x16x32_f16(false, ah, false, b, (short)0, aRH, false, false);
    b = *(const v16h*)(bz_hh + kb);
    aZH = __builtin_amdgcn_wmma_f32_16x16x32_f16(false, ah, false, b, (short)0, aZH, false, false);
    b = *(const v16h*)(bn_hh + kb);
    aNH = __builtin_amdgcn_wmma_f32_16x16x32_f16(false, ah, false, b, (short)0, aNH, false, false);
  }

  // GRU epilogue. C layout: VGPR r -> (M = r + 8*lh, N = laneM)
  const int s = s0 + laneM;
  const float bir = bih[t * G3 + s];
  const float biz = bih[t * G3 + SH + s];
  const float bin = bih[t * G3 + 2 * SH + s];
  const float bhr = bhh[t * G3 + s];
  const float bhz = bhh[t * G3 + SH + s];
  const float bhn = bhh[t * G3 + 2 * SH + s];
  for (int r = 0; r < 8; ++r) {
    const int m = m0 + r + lh * 8;
    const float ho = hf32[(size_t)m * SH + s];
    const float rr = 1.0f / (1.0f + __builtin_amdgcn_exp2f(-(aRX[r] + aRH[r] + bir + bhr) * 1.4426950408889634f));
    const float zz = 1.0f / (1.0f + __builtin_amdgcn_exp2f(-(aZX[r] + aZH[r] + biz + bhz) * 1.4426950408889634f));
    const float narg = aNX[r] + bin + rr * (aNH[r] + bhn);
    const float nn = 1.0f - 2.0f / (__builtin_amdgcn_exp2f(2.0f * narg * 1.4426950408889634f) + 1.0f);
    hnext[((size_t)m * T_N + t) * SH + s] = (1.0f - zz) * nn + zz * ho;
  }
}

// ---------- write_key[m,t,k] = sum_s hnext[m,t,s] * w_write[t,s,k] ----------
__global__ void wkey_kernel(const float* hnext, const float* wwrite, float* wkey) {
  int gid = blockIdx.x * blockDim.x + threadIdx.x;  // (m*8+t)*16 + k
  int k = gid & 15;
  int pair = gid >> 4;
  int t = pair & 7;
  const float* hn = hnext + (size_t)pair * SH;
  const float* ww = wwrite + (size_t)t * SH * 16 + k;
  float acc = 0.f;
  for (int s = 0; s < SH; ++s) acc += hn[s] * ww[s * 16];
  wkey[gid] = acc;
}

// ---------- logits[m,t] = sum_k h_read[m,k] * write_key[m,t,k] ----------
__global__ void logits_kernel(const float* wkey, const float* hread, float* logits) {
  int pair = blockIdx.x * blockDim.x + threadIdx.x;  // m*8+t
  int m = pair >> 3;
  float acc = 0.f;
  for (int k = 0; k < 16; ++k) acc += hread[m * 16 + k] * wkey[pair * 16 + k];
  logits[pair] = acc;
}

// ---------- gumbel-hard: one-hot(argmax(logits + gumbel)), JAX threefry key [0,42] ----------
__global__ void gumbel_argmax_kernel(const float* logits, float* att_out, int* tstar) {
  int m = blockIdx.x * blockDim.x + threadIdx.x;
  if (m >= M_TOTAL) return;
  const float tiny = 1.17549435e-38f;
  const unsigned key0 = 0u, key1 = 42u;
  const unsigned key2 = key0 ^ key1 ^ 0x1BD11BDAu;
  const int rots[5][4] = {{13, 15, 26, 6}, {17, 29, 16, 24}, {13, 15, 26, 6},
                          {17, 29, 16, 24}, {13, 15, 26, 6}};
  const unsigned inj0[5] = {key1, key2, key0, key1, key2};
  const unsigned inj1[5] = {key2 + 1u, key0 + 2u, key1 + 3u, key2 + 4u, key0 + 5u};

  float best = -3.0e38f;
  int bt = 0;
  for (int t = 0; t < T_N; ++t) {
    unsigned flat = (unsigned)(m * T_N + t);          // iota counter position
    unsigned c0, c1;
    if (flat < 16384u) { c0 = flat; c1 = flat + 16384u; }
    else               { c0 = flat - 16384u; c1 = flat; }
    // threefry2x32, 5 groups of 4 rounds, inlined
    unsigned x0 = c0 + key0, x1 = c1 + key1;
    for (int gr = 0; gr < 5; ++gr) {
      for (int rr = 0; rr < 4; ++rr) {
        int ro = rots[gr][rr];
        x0 += x1;
        x1 = (x1 << ro) | (x1 >> (32 - ro));
        x1 ^= x0;
      }
      x0 += inj0[gr];
      x1 += inj1[gr];
    }
    unsigned bits = (flat < 16384u) ? x0 : x1;
    // jax uniform: bitcast((bits>>9)|0x3F800000)-1  ==  (float)(bits>>9) * 2^-23 (exact)
    float u01 = (float)(bits >> 9) * 1.1920929e-7f;   // 2^-23
    float u = u01 * (1.0f - tiny) + tiny;
    if (u < tiny) u = tiny;
    float l1 = __builtin_amdgcn_logf(u) * 0.6931471805599453f;        // ln(u) < 0
    float g = -(__builtin_amdgcn_logf(-l1) * 0.6931471805599453f);    // -ln(-ln u)
    float score = logits[flat] + g;
    if (score > best) { best = score; bt = t; }       // strict > == argmax tie rule
  }
  tstar[m] = bt;
  for (int t = 0; t < T_N; ++t) att_out[m * T_N + t] = (t == bt) ? 1.0f : 0.0f;
}

// ---------- hout[m, :] = hnext[m, tstar[m], :] ----------
__global__ void select_kernel(const float* hnext, const int* tstar, float* hout) {
  int i = blockIdx.x * blockDim.x + threadIdx.x;   // m*512 + s
  int m = i >> 9;
  int s = i & 511;
  hout[i] = hnext[((size_t)m * T_N + tstar[m]) * SH + s];
}

// ---------- launch ----------
extern "C" void kernel_launch(void* const* d_in, const int* in_sizes, int n_in,
                              void* d_out, int out_size, void* d_ws, size_t ws_size,
                              hipStream_t stream) {
  const float* x      = (const float*)d_in[0];
  const float* h      = (const float*)d_in[1];
  const float* Wih    = (const float*)d_in[2];
  const float* Whh    = (const float*)d_in[3];
  const float* bih    = (const float*)d_in[4];
  const float* bhh    = (const float*)d_in[5];
  const float* wread  = (const float*)d_in[6];
  const float* wwrite = (const float*)d_in[7];

  float* out  = (float*)d_out;
  float* hout = out;                                // 4096*512
  float* att  = out + (size_t)M_TOTAL * SH;         // 4096*8

  char* w = (char*)d_ws;
  unsigned short* xp   = (unsigned short*)w; w += (size_t)M_TOTAL * DIN * 2;
  unsigned short* hp   = (unsigned short*)w; w += (size_t)M_TOTAL * SH * 2;
  unsigned short* Wihb = (unsigned short*)w; w += (size_t)T_N * G3 * KDIM * 2;
  unsigned short* Whhb = (unsigned short*)w; w += (size_t)T_N * G3 * KDIM * 2;
  float* hnext = (float*)w;  w += (size_t)M_TOTAL * T_N * SH * 4;
  float* wkey  = (float*)w;  w += (size_t)M_TOTAL * T_N * 16 * 4;
  float* hrd   = (float*)w;  w += (size_t)M_TOTAL * 16 * 4;
  float* logit = (float*)w;  w += (size_t)M_TOTAL * T_N * 4;
  int*   tst   = (int*)w;

  const int nX = M_TOTAL * DIN;                     // 2,097,152
  const int nW = T_N * G3 * KDIM;                   // 6,291,456
  cvt_pack_a_kernel<<<(nX + 255) / 256, 256, 0, stream>>>(x, xp, nX);
  cvt_pack_a_kernel<<<(nX + 255) / 256, 256, 0, stream>>>(h, hp, nX);
  cvt_f16_kernel<<<(nW + 255) / 256, 256, 0, stream>>>(Wih, Wihb, nW);
  cvt_f16_kernel<<<(nW + 255) / 256, 256, 0, stream>>>(Whh, Whhb, nW);

  hread_kernel<<<(M_TOTAL * 16) / 256, 256, 0, stream>>>(h, wread, hrd);

  SharedBlockGRU_19636590477500_kernel<<<(M_TOTAL / 16) * T_N * 8, 128, 0, stream>>>(
      xp, hp, Wihb, Whhb, bih, bhh, h, hnext);

  wkey_kernel<<<(M_TOTAL * T_N * 16) / 256, 256, 0, stream>>>(hnext, wwrite, wkey);
  logits_kernel<<<(M_TOTAL * T_N) / 256, 256, 0, stream>>>(wkey, hrd, logit);

  gumbel_argmax_kernel<<<(M_TOTAL + 255) / 256, 256, 0, stream>>>(logit, att, tst);

  select_kernel<<<(M_TOTAL * SH) / 256, 256, 0, stream>>>(hnext, tst, hout);
}
